// MMTransformerBlock_69088843924226
// MI455X (gfx1250) — matrix operand order repaired and checked by reading, not compiled, not run on hardware
//
#include <hip/hip_runtime.h>

typedef __bf16 bf16_t;
typedef __attribute__((ext_vector_type(16))) __bf16 v16bf;
typedef __attribute__((ext_vector_type(8)))  __bf16 v8bf;
typedef __attribute__((ext_vector_type(8)))  float  v8f;

#define B_  2
#define S_  2048
#define D_  1024
#define H_  16
#define G_  4
#define HD_ 64
#define FF_ 4096
#define M_  (B_*S_)   // 4096 token rows

#if defined(__has_builtin)
#if __has_builtin(__builtin_amdgcn_sched_barrier)
#define SCHED_BARRIER() __builtin_amdgcn_sched_barrier(0)
#endif
#endif
#ifndef SCHED_BARRIER
#define SCHED_BARRIER()
#endif

// ---------------------------------------------------------------- WMMA helpers

__device__ inline v8f wmma_bf16(v16bf a, v16bf b, v8f c) {
  // D = A(16x32) x B(32x16) + C, f32 accumulate
  return __builtin_amdgcn_wmma_f32_16x16x32_bf16(false, a, false, b, (short)0, c,
                                                 false, false);
}

__device__ inline v8f vzero8f() {
  v8f r;
#pragma unroll
  for (int i = 0; i < 8; ++i) r[i] = 0.f;
  return r;
}

// A-fragment (16x32 bf16, MxK): lane holds row m = lane%16; halves cover
// k = grp*8 + {0..7} (chunk0) and 16 + grp*8 + {0..7} (chunk1).
__device__ inline v16bf load_afrag(const bf16_t* __restrict__ p, int grp) {
  v8bf c0 = *(const v8bf*)(p + grp * 8);
  v8bf c1 = *(const v8bf*)(p + 16 + grp * 8);
  v16bf r;
#pragma unroll
  for (int i = 0; i < 8; ++i) { r[i] = c0[i]; r[8 + i] = c1[i]; }
  return r;
}

// B-fragment (32x16 bf16, KxN) from [N,K] row-major storage: lane holds
// column n = lane%16; halves cover k = grp*8 + {0..7} and 16 + grp*8 + {0..7}.
__device__ inline v16bf load_bfrag(const bf16_t* __restrict__ p, int grp) {
  v8bf c0 = *(const v8bf*)(p + grp * 8);
  v8bf c1 = *(const v8bf*)(p + 16 + grp * 8);
  v16bf r;
#pragma unroll
  for (int i = 0; i < 8; ++i) { r[i] = c0[i]; r[8 + i] = c1[i]; }
  return r;
}

// ---------------------------------------------------------------- elementwise

__global__ __launch_bounds__(256) void cvt_bf16_kernel(
    const float* __restrict__ src, bf16_t* __restrict__ dst, int n) {
  int t = blockIdx.x * 256 + threadIdx.x;
  if (t < n) dst[t] = (bf16_t)src[t];
}

// one block per row of D_ floats; bf16 normalized output
__global__ __launch_bounds__(256) void rmsnorm_kernel(
    const float* __restrict__ x, const float* __restrict__ w,
    bf16_t* __restrict__ out) {
  __shared__ float red[256];
  const int row = blockIdx.x;
  const float* xr = x + (size_t)row * D_;
  float vals[4];
  float ss = 0.f;
#pragma unroll
  for (int i = 0; i < 4; ++i) {
    vals[i] = xr[threadIdx.x + i * 256];
    ss += vals[i] * vals[i];
  }
  red[threadIdx.x] = ss;
  __syncthreads();
  for (int s = 128; s > 0; s >>= 1) {
    if (threadIdx.x < (unsigned)s) red[threadIdx.x] += red[threadIdx.x + s];
    __syncthreads();
  }
  const float inv = rsqrtf(red[0] * (1.0f / D_) + 1e-6f);
#pragma unroll
  for (int i = 0; i < 4; ++i) {
    int c = threadIdx.x + i * 256;
    out[(size_t)row * D_ + c] = (bf16_t)(vals[i] * inv * w[c]);
  }
}

// RoPE: each thread handles one (j, j+32) pair. raw: [b][s][heads*64] f32.
__global__ __launch_bounds__(256) void rope_kernel(
    const float* __restrict__ raw, const float* __restrict__ cosb,
    const float* __restrict__ sinb, bf16_t* __restrict__ out, int heads) {
  int t = blockIdx.x * 256 + threadIdx.x;
  int j = t & 31;
  int hh = (t >> 5) % heads;
  int s = (t / (32 * heads)) % S_;
  int b = t / (32 * heads * S_);
  size_t base = ((size_t)(b * S_ + s)) * (heads * HD_) + hh * HD_;
  float x1 = raw[base + j];
  float x2 = raw[base + 32 + j];
  float c1 = cosb[s * HD_ + j],      s1 = sinb[s * HD_ + j];
  float c2 = cosb[s * HD_ + 32 + j], s2 = sinb[s * HD_ + 32 + j];
  out[base + j]      = (bf16_t)(x1 * c1 - x2 * s1);
  out[base + 32 + j] = (bf16_t)(x2 * c2 + x1 * s2);
}

// v_raw [b][s][G*64] f32  ->  vt [b][g][hd][S] bf16 (so P·V B-frags are contiguous)
__global__ __launch_bounds__(256) void vtrans_kernel(
    const float* __restrict__ vraw, bf16_t* __restrict__ vt) {
  int t = blockIdx.x * 256 + threadIdx.x;   // enumerates vraw linearly
  int c = t & 255;          // g*64 + hd
  int s = (t >> 8) & (S_ - 1);
  int b = t >> 19;          // 2048*256 = 2^19
  int g = c >> 6, hd = c & 63;
  vt[(((size_t)(b * G_ + g)) * HD_ + hd) * S_ + s] = (bf16_t)vraw[t];
}

// ---------------------------------------------------------------- WMMA GEMM
// C[M,N] = A[M,K] (bf16, row-major) x B0[N,K]^T (bf16, row-major)
// EPI 0: f32 out            EPI 1: f32 out + residual R[M,N]
// EPI 2: bf16 out = silu(A·B0^T) * (A·B1^T)   (fused SwiGLU)
// Block: 8 waves as 4(M) x 2(N). Wave tile: MT*16 (M) x 64 (N).
// Block tile: (4*MT*16) x 128. K multiple of 64, double-buffered K loop with
// sched_barriers pinning the load/compute phases (loads of phase i+1 issue
// before the WMMAs of phase i, so waits become partial).

template <int EPI, int MT>
__global__ __launch_bounds__(256) void gemm_kernel(
    const bf16_t* __restrict__ A, const bf16_t* __restrict__ B0,
    const bf16_t* __restrict__ B1, const float* __restrict__ R,
    void* __restrict__ OutP, int M, int N, int K) {
  const int lane = threadIdx.x & 31;
  const int w = threadIdx.x >> 5;
  const int lm = lane & 15, grp = lane >> 4;
  const int m0 = blockIdx.x * (4 * MT * 16) + (w & 3) * (MT * 16);
  const int n0 = blockIdx.y * 128 + (w >> 2) * 64;

  v8f acc[MT][4], acc2[MT][4];
#pragma unroll
  for (int t = 0; t < MT; ++t)
#pragma unroll
    for (int j = 0; j < 4; ++j) { acc[t][j] = vzero8f(); acc2[t][j] = vzero8f(); }

  const bf16_t* arow[MT];
  const bf16_t* brow[4];
  const bf16_t* brow2[4];
#pragma unroll
  for (int t = 0; t < MT; ++t) arow[t] = A + (size_t)(m0 + t * 16 + lm) * K;
#pragma unroll
  for (int j = 0; j < 4; ++j) {
    brow[j] = B0 + (size_t)(n0 + j * 16 + lm) * K;
    if (EPI == 2) brow2[j] = B1 + (size_t)(n0 + j * 16 + lm) * K;
  }

  // double buffers
  v16bf aX[MT], bX[4], b2X[4];   // even phase
  v16bf aY[MT], bY[4], b2Y[4];   // odd phase

  // preload phase-X fragments for kk = 0
#pragma unroll
  for (int t = 0; t < MT; ++t) aX[t] = load_afrag(arow[t], grp);
#pragma unroll
  for (int j = 0; j < 4; ++j) {
    bX[j] = load_bfrag(brow[j], grp);
    if (EPI == 2) b2X[j] = load_bfrag(brow2[j], grp);
  }
  SCHED_BARRIER();

  for (int kk = 0; kk < K; kk += 64) {
    // issue loads for kk+32 (phase Y) — independent of phase-X WMMAs
#pragma unroll
    for (int t = 0; t < MT; ++t) aY[t] = load_afrag(arow[t] + kk + 32, grp);
#pragma unroll
    for (int j = 0; j < 4; ++j) {
      bY[j] = load_bfrag(brow[j] + kk + 32, grp);
      if (EPI == 2) b2Y[j] = load_bfrag(brow2[j] + kk + 32, grp);
    }
    SCHED_BARRIER();
    // compute phase X (kk): operands were loaded one full phase ago
#pragma unroll
    for (int t = 0; t < MT; ++t)
#pragma unroll
      for (int j = 0; j < 4; ++j) {
        acc[t][j] = wmma_bf16(aX[t], bX[j], acc[t][j]);
        if (EPI == 2) acc2[t][j] = wmma_bf16(aX[t], b2X[j], acc2[t][j]);
      }
    SCHED_BARRIER();
    // issue loads for kk+64 (phase X of next iter), guarded at the tail
    if (kk + 64 < K) {
#pragma unroll
      for (int t = 0; t < MT; ++t) aX[t] = load_afrag(arow[t] + kk + 64, grp);
#pragma unroll
      for (int j = 0; j < 4; ++j) {
        bX[j] = load_bfrag(brow[j] + kk + 64, grp);
        if (EPI == 2) b2X[j] = load_bfrag(brow2[j] + kk + 64, grp);
      }
    }
    SCHED_BARRIER();
    // compute phase Y (kk+32)
#pragma unroll
    for (int t = 0; t < MT; ++t)
#pragma unroll
      for (int j = 0; j < 4; ++j) {
        acc[t][j] = wmma_bf16(aY[t], bY[j], acc[t][j]);
        if (EPI == 2) acc2[t][j] = wmma_bf16(aY[t], b2Y[j], acc2[t][j]);
      }
    SCHED_BARRIER();
  }

#pragma unroll
  for (int t = 0; t < MT; ++t)
#pragma unroll
    for (int j = 0; j < 4; ++j)
#pragma unroll
      for (int v = 0; v < 8; ++v) {
        size_t row = (size_t)(m0 + t * 16 + v + 8 * grp);
        size_t col = (size_t)(n0 + j * 16 + lm);
        size_t idx = row * (size_t)N + col;
        float a = acc[t][j][v];
        if (EPI == 0) {
          ((float*)OutP)[idx] = a;
        } else if (EPI == 1) {
          ((float*)OutP)[idx] = a + R[idx];
        } else {
          float gate = a / (1.f + __expf(-a));   // silu
          ((bf16_t*)OutP)[idx] = (bf16_t)(gate * acc2[t][j][v]);
        }
      }
}

// ---------------------------------------------------------------- attention
// One wave per (b, h, 16-query tile). Register flash attention.
// S^T = K·Q^T so each lane owns query column (lane%16); the S^T D-layout maps
// directly onto the P A-fragment layout (keys grp*8+v / 16+grp*8+v).
// All 8 fragment loads (4 K, 4 V) of a key block are issued as one group (pinned by a
// sched_barrier) so the first WMMA needs only a partial wait and the V
// fragments arrive during the softmax VALU work.

__global__ __launch_bounds__(256) void attn_kernel(
    const bf16_t* __restrict__ Qb,   // [B,S,H*64] bf16 (post-RoPE)
    const bf16_t* __restrict__ Kb,   // [B,S,G*64] bf16 (post-RoPE)
    const bf16_t* __restrict__ Vt,   // [B,G,64,S] bf16
    bf16_t* __restrict__ Ob) {       // [B,S,H*64] bf16
  const int lane = threadIdx.x & 31;
  const int gw = blockIdx.x * 8 + (threadIdx.x >> 5);  // 4096 waves total
  const int QT = S_ / 16;                              // 128
  const int qt = gw % QT;
  const int h = (gw / QT) % H_;
  const int b = gw / (QT * H_);
  const int g = h >> 2;
  const int q0 = qt * 16;
  const int lm = lane & 15, grp = lane >> 4;
  const int qi = q0 + lm;   // this lane's query index

  // Q loaded once as two B-fragments (hd 0..31, 32..63)
  const bf16_t* qrow = Qb + ((size_t)(b * S_ + q0 + lm)) * (H_ * HD_) + h * HD_;
  v16bf qb0 = load_bfrag(qrow, grp);
  v16bf qb1 = load_bfrag(qrow + 32, grp);

  const bf16_t* kbase = Kb + (size_t)b * S_ * (G_ * HD_) + g * HD_;
  const bf16_t* vbase = Vt + ((size_t)(b * G_ + g)) * HD_ * S_;

  float rmax = -1e30f, rsum = 0.f;
  v8f ov[4];
#pragma unroll
  for (int t = 0; t < 4; ++t) ov[t] = vzero8f();

  for (int kb = 0; kb < q0 + 16; kb += 32) {
    // ---- issue ALL loads for this key block up front (pinned) ----
    const bf16_t* kr0 = kbase + (size_t)(kb + lm) * (G_ * HD_);
    const bf16_t* kr1 = kbase + (size_t)(kb + 16 + lm) * (G_ * HD_);
    v16bf ka00 = load_afrag(kr0, grp);
    v16bf ka01 = load_afrag(kr0 + 32, grp);
    v16bf ka10 = load_afrag(kr1, grp);
    v16bf ka11 = load_afrag(kr1 + 32, grp);
    v16bf vb[4];
#pragma unroll
    for (int t = 0; t < 4; ++t)
      vb[t] = load_bfrag(vbase + (size_t)(t * 16 + lm) * S_ + kb, grp);
    SCHED_BARRIER();

    // ---- S^T tiles: keys kb..kb+15 (st0) and kb+16..kb+31 (st1) ----
    v8f st0 = vzero8f(), st1 = vzero8f();
    st0 = wmma_bf16(ka00, qb0, st0);
    st1 = wmma_bf16(ka10, qb0, st1);
    st0 = wmma_bf16(ka01, qb1, st0);
    st1 = wmma_bf16(ka11, qb1, st1);
    SCHED_BARRIER();

    // ---- scale + causal mask + block max ----
    float bm = -1e30f;
#pragma unroll
    for (int v = 0; v < 8; ++v) {
      int k0 = kb + grp * 8 + v;
      int k1 = kb + 16 + grp * 8 + v;
      float s0 = st0[v] * 0.125f;
      float s1 = st1[v] * 0.125f;
      if (k0 > qi) s0 = -1e30f;
      if (k1 > qi) s1 = -1e30f;
      st0[v] = s0; st1[v] = s1;
      bm = fmaxf(bm, fmaxf(s0, s1));
    }
    bm = fmaxf(bm, __shfl_xor(bm, 16, 32));   // full 32-key block max per query
    float mnew = fmaxf(rmax, bm);
    float fac = __expf(rmax - mnew);

    // ---- P = exp(S - m); S^T D-layout == P A-fragment layout ----
    float ls = 0.f;
    v16bf pa;
#pragma unroll
    for (int v = 0; v < 8; ++v) {
      float p0 = __expf(st0[v] - mnew);
      float p1 = __expf(st1[v] - mnew);
      ls += p0 + p1;
      pa[v] = (bf16_t)p0;
      pa[8 + v] = (bf16_t)p1;
    }
    ls += __shfl_xor(ls, 16, 32);
    rsum = rsum * fac + ls;
    rmax = mnew;

    // ---- O rescale (lane's rows are queries m = v + 8*grp) + P·V ----
    v8f fv;
#pragma unroll
    for (int v = 0; v < 8; ++v) fv[v] = __shfl(fac, grp * 8 + v, 32);
#pragma unroll
    for (int t = 0; t < 4; ++t) {
#pragma unroll
      for (int v = 0; v < 8; ++v) ov[t][v] *= fv[v];
      ov[t] = wmma_bf16(pa, vb[t], ov[t]);
    }
    SCHED_BARRIER();
  }

  // normalize by row sum, write [b,s,h*64+hd]
  float inv = 1.f / rsum;
  v8f iv;
#pragma unroll
  for (int v = 0; v < 8; ++v) iv[v] = __shfl(inv, grp * 8 + v, 32);
#pragma unroll
  for (int t = 0; t < 4; ++t)
#pragma unroll
    for (int v = 0; v < 8; ++v) {
      int m = v + 8 * grp;
      int n = t * 16 + lm;
      Ob[((size_t)(b * S_ + q0 + m)) * (H_ * HD_) + h * HD_ + n] =
          (bf16_t)(ov[t][v] * iv[v]);
    }
}

// ---------------------------------------------------------------- launch

extern "C" void kernel_launch(void* const* d_in, const int* in_sizes, int n_in,
                              void* d_out, int out_size, void* d_ws,
                              size_t ws_size, hipStream_t stream) {
  (void)in_sizes; (void)n_in; (void)out_size; (void)ws_size;
  const float* x    = (const float*)d_in[0];
  // d_in[1] = mask (unused: causality computed analytically)
  const float* cosb = (const float*)d_in[2];
  const float* sinb = (const float*)d_in[3];
  const float* n1w  = (const float*)d_in[4];
  const float* n2w  = (const float*)d_in[5];
  const float* wq   = (const float*)d_in[6];
  const float* wk   = (const float*)d_in[7];
  const float* wv   = (const float*)d_in[8];
  const float* wo   = (const float*)d_in[9];
  const float* w1   = (const float*)d_in[10];
  const float* w3   = (const float*)d_in[11];
  const float* w2   = (const float*)d_in[12];

  char* p = (char*)d_ws;
  auto alloc = [&](size_t bytes) -> void* {
    void* r = (void*)p;
    p += (bytes + 255) & ~(size_t)255;
    return r;
  };
  bf16_t* wq_bf  = (bf16_t*)alloc((size_t)D_ * D_ * 2);
  bf16_t* wk_bf  = (bf16_t*)alloc((size_t)G_ * HD_ * D_ * 2);
  bf16_t* wv_bf  = (bf16_t*)alloc((size_t)G_ * HD_ * D_ * 2);
  bf16_t* wo_bf  = (bf16_t*)alloc((size_t)D_ * D_ * 2);
  bf16_t* w1_bf  = (bf16_t*)alloc((size_t)FF_ * D_ * 2);
  bf16_t* w3_bf  = (bf16_t*)alloc((size_t)FF_ * D_ * 2);
  bf16_t* w2_bf  = (bf16_t*)alloc((size_t)D_ * FF_ * 2);
  bf16_t* xn1_bf = (bf16_t*)alloc((size_t)M_ * D_ * 2);
  float*  q_raw  = (float*) alloc((size_t)M_ * D_ * 4);
  float*  k_raw  = (float*) alloc((size_t)M_ * G_ * HD_ * 4);
  float*  v_raw  = (float*) alloc((size_t)M_ * G_ * HD_ * 4);
  bf16_t* q_bf   = (bf16_t*)alloc((size_t)M_ * D_ * 2);
  bf16_t* k_bf   = (bf16_t*)alloc((size_t)M_ * G_ * HD_ * 2);
  bf16_t* vt_bf  = (bf16_t*)alloc((size_t)M_ * G_ * HD_ * 2);
  bf16_t* o_bf   = (bf16_t*)alloc((size_t)M_ * D_ * 2);
  float*  skip   = (float*) alloc((size_t)M_ * D_ * 4);
  bf16_t* xn2_bf = (bf16_t*)alloc((size_t)M_ * D_ * 2);
  bf16_t* hb_bf  = (bf16_t*)alloc((size_t)M_ * FF_ * 2);

  const dim3 blk(256);
  auto cvt = [&](const float* s, bf16_t* d, int n) {
    cvt_bf16_kernel<<<dim3((n + 255) / 256), blk, 0, stream>>>(s, d, n);
  };
  cvt(wq, wq_bf, D_ * D_);
  cvt(wk, wk_bf, G_ * HD_ * D_);
  cvt(wv, wv_bf, G_ * HD_ * D_);
  cvt(wo, wo_bf, D_ * D_);
  cvt(w1, w1_bf, FF_ * D_);
  cvt(w3, w3_bf, FF_ * D_);
  cvt(w2, w2_bf, D_ * FF_);

  // xn1 = rmsnorm(x)
  rmsnorm_kernel<<<dim3(M_), blk, 0, stream>>>(x, n1w, xn1_bf);

  // QKV projections (MT=4 -> block tile 256x128)
  gemm_kernel<0, 4><<<dim3(M_ / 256, D_ / 128), blk, 0, stream>>>(
      xn1_bf, wq_bf, nullptr, nullptr, q_raw, M_, D_, D_);
  gemm_kernel<0, 4><<<dim3(M_ / 256, (G_ * HD_) / 128), blk, 0, stream>>>(
      xn1_bf, wk_bf, nullptr, nullptr, k_raw, M_, G_ * HD_, D_);
  gemm_kernel<0, 4><<<dim3(M_ / 256, (G_ * HD_) / 128), blk, 0, stream>>>(
      xn1_bf, wv_bf, nullptr, nullptr, v_raw, M_, G_ * HD_, D_);

  // RoPE (q, k) and V transpose, all to bf16
  rope_kernel<<<dim3(B_ * S_ * H_ * 32 / 256), blk, 0, stream>>>(
      q_raw, cosb, sinb, q_bf, H_);
  rope_kernel<<<dim3(B_ * S_ * G_ * 32 / 256), blk, 0, stream>>>(
      k_raw, cosb, sinb, k_bf, G_);
  vtrans_kernel<<<dim3(B_ * S_ * G_ * HD_ / 256), blk, 0, stream>>>(v_raw, vt_bf);

  // flash attention: B*H*(S/16) = 4096 waves, 8 per block
  attn_kernel<<<dim3(B_ * H_ * (S_ / 16) / 8), blk, 0, stream>>>(
      q_bf, k_bf, vt_bf, o_bf);

  // skipped = x + o @ wo^T
  gemm_kernel<1, 4><<<dim3(M_ / 256, D_ / 128), blk, 0, stream>>>(
      o_bf, wo_bf, nullptr, x, skip, M_, D_, D_);

  // xn2 = rmsnorm(skipped)
  rmsnorm_kernel<<<dim3(M_), blk, 0, stream>>>(skip, n2w, xn2_bf);

  // hb = silu(xn2 @ w1^T) * (xn2 @ w3^T)  (fused dual GEMM, MT=2)
  gemm_kernel<2, 2><<<dim3(M_ / 128, FF_ / 128), blk, 0, stream>>>(
      xn2_bf, w1_bf, w3_bf, nullptr, hb_bf, M_, FF_, D_);

  // out = hb @ w2^T + skipped
  gemm_kernel<1, 4><<<dim3(M_ / 256, D_ / 128), blk, 0, stream>>>(
      hb_bf, w2_bf, nullptr, skip, (float*)d_out, M_, D_, FF_);
}
